// PredRNN_53953379173237
// MI455X (gfx1250) — compile-verified, matrix operand order checked by reading, not run on hardware
//
#include <hip/hip_runtime.h>

#define LNUM 4
#define BSZ  128
#define TSEQ 512
#define DIM  512
#define HID  512
#define ODIM 512
#define FEPS 0.1f

typedef unsigned short u16;
typedef __attribute__((ext_vector_type(16))) __bf16 v16bf;
typedef __attribute__((ext_vector_type(8)))  float  v8f;

// ---------- helpers ----------

__device__ __forceinline__ u16 f2bf(float f) {
    unsigned u = __float_as_uint(f);
    unsigned r = 0x7fffu + ((u >> 16) & 1u);   // round-to-nearest-even
    return (u16)((u + r) >> 16);
}
__device__ __forceinline__ float bf2f(u16 h) {
    return __uint_as_float(((unsigned)h) << 16);
}
__device__ __forceinline__ float sigmoidf_(float x) {
    return 1.0f / (1.0f + __expf(-x));
}
__device__ __forceinline__ float tanhf_(float x) {
    float e = __expf(-2.0f * __builtin_fabsf(x));
    float r = (1.0f - e) / (1.0f + e);
    return __builtin_copysignf(r, x);
}

// A-fragment: 16x32 bf16 tile, per-lane 32B contiguous in fragment order
__device__ __forceinline__ v16bf afrag(const u16* p, int kt, int lane) {
    return *reinterpret_cast<const v16bf*>(p + (kt << 9) + (lane << 4));
}
// B-fragment: 32x16 bf16 tile from packed weights
__device__ __forceinline__ v16bf bfrag(const u16* w, long frag, int lane) {
    return *reinterpret_cast<const v16bf*>(w + (frag << 9) + ((long)lane << 4));
}

#define WMMA(acc, a, bm) \
    acc = __builtin_amdgcn_wmma_f32_16x16x32_bf16(false, (a), false, (bm), (short)0, (acc), false, false)

// two adjacent N-tiles (nt, nt+1) share the A fragment; their B frags are
// contiguous in the packed stream -> 2 independent load+WMMA chains per gate
#define GATE2(a0, a1, av, w, fb) do {              \
    v16bf b0_ = bfrag((w), (fb),     lane);        \
    v16bf b1_ = bfrag((w), (fb) + 1, lane);        \
    WMMA(a0, (av), b0_);                           \
    WMMA(a1, (av), b1_); } while (0)

// Write a C/D fragment (column tile nt) into A-fragment layout.
// C layout: VGPR r -> row M = r + (lane>=16 ? 8:0), col = nt*16 + (lane&15).
__device__ __forceinline__ void store_state(u16* p, int nt, int lane, const v8f v) {
    const int cc = lane & 15;
    const int hi = (lane >> 4) << 3;
    u16* q = p + ((nt >> 1) << 9) + ((nt & 1) << 8) + cc;
    #pragma unroll
    for (int r = 0; r < 8; ++r) q[(r + hi) << 4] = f2bf(v[r]);
}
__device__ __forceinline__ v8f load_state(const u16* p, int nt, int lane) {
    const int cc = lane & 15;
    const int hi = (lane >> 4) << 3;
    const u16* q = p + ((nt >> 1) << 9) + ((nt & 1) << 8) + cc;
    v8f v;
    #pragma unroll
    for (int r = 0; r < 8; ++r) v[r] = bf2f(q[(r + hi) << 4]);
    return v;
}

// ---------- packing kernels (fp32 -> bf16, WMMA fragment order) ----------

__global__ void pack_w_kernel(const float* __restrict__ src, u16* __restrict__ dst,
                              int K, int N, long total) {
    long idx = (long)blockIdx.x * blockDim.x + threadIdx.x;
    if (idx >= total) return;
    long per  = (long)K * N;
    long mat  = idx / per;
    long p    = idx - mat * per;
    int  frag = (int)(p >> 9);
    int  lane = (int)((p >> 4) & 31);
    int  e    = (int)(p & 15);
    int  ntl  = N >> 4;
    int  kt   = frag / ntl;
    int  nnt  = frag - kt * ntl;
    int  k    = (kt << 5) + ((lane >> 4) << 4) + e;
    int  n    = (nnt << 4) + (lane & 15);
    dst[idx] = f2bf(src[mat * per + (long)k * N + n]);
}

__global__ void pack_x_kernel(const float* __restrict__ x, u16* __restrict__ xp, long total) {
    long idx = (long)blockIdx.x * blockDim.x + threadIdx.x;
    if (idx >= total) return;
    int tile = (int)(idx >> 13);            // (bt*T + t)
    int q    = (int)(idx & 8191);
    int bt   = tile >> 9;                   // / TSEQ
    int t    = tile & 511;
    int kt   = q >> 9;
    int lane = (q >> 4) & 31;
    int e    = q & 15;
    int b    = (bt << 4) + (lane & 15);
    int d    = (kt << 5) + ((lane >> 4) << 4) + e;
    xp[idx] = f2bf(x[(((long)b * TSEQ) + t) * DIM + d]);
}

// ---------- persistent ST-LSTM recurrence: 1 block = 16 batch rows ----------
// 512 threads = 16 waves; each wave owns two adjacent 16-column tiles of H.

__global__ void __launch_bounds__(512)
stlstm_kernel(const u16* __restrict__ wxp, const u16* __restrict__ whp,
              const u16* __restrict__ wmp, const u16* __restrict__ wcop,
              const u16* __restrict__ wmop, const u16* __restrict__ w11p,
              const float* __restrict__ bvec, const u16* __restrict__ xpack,
              u16* __restrict__ hall) {
    __shared__ __align__(32) u16 hbuf[LNUM * 8192];   // h state per layer (A-frag layout)
    __shared__ __align__(32) u16 mbuf[LNUM * 8192];   // m state per layer
    __shared__ __align__(32) u16 cnew[8192];          // c_new scratch (per layer, reused)
    __shared__ __align__(32) u16 xbuf[8192];          // staged x_t tile

    const int tid  = threadIdx.x;
    const int lane = tid & 31;
    const int wv   = tid >> 5;                        // 0..15
    const int nt0  = wv << 1;                         // first column tile
    const int nt1  = nt0 | 1;                         // second column tile
    const int bt   = blockIdx.x;
    const int col0 = (nt0 << 4) + (lane & 15);        // col1 = col0 + 16

    for (int i = tid; i < LNUM * 8192; i += 512) { hbuf[i] = 0; mbuf[i] = 0; }
    __syncthreads();

    v8f c_reg[LNUM][2];
    #pragma unroll
    for (int l = 0; l < LNUM; ++l) {
        c_reg[l][0] = (v8f){0,0,0,0,0,0,0,0};
        c_reg[l][1] = (v8f){0,0,0,0,0,0,0,0};
    }

    for (int t = 0; t < TSEQ; ++t) {
        // stage x_t fragment tile (16KB) via CDNA5 async global->LDS copy
        {
            unsigned long long g =
                (unsigned long long)(xpack + ((size_t)(bt * TSEQ + t) << 13)) +
                ((unsigned)tid << 4);
            unsigned l0 = (unsigned)(size_t)xbuf + ((unsigned)tid << 4);
            asm volatile("global_load_async_to_lds_b128 %0, %1, off"
                         :: "v"(l0), "v"(g) : "memory");
            asm volatile("global_load_async_to_lds_b128 %0, %1, off offset:8192"
                         :: "v"(l0), "v"(g) : "memory");
            asm volatile("s_wait_asynccnt 0x0" ::: "memory");
        }
        __syncthreads();

        #pragma unroll
        for (int l = 0; l < LNUM; ++l) {
            const u16* inp = (l == 0) ? xbuf : (hbuf + ((l - 1) << 13));
            const u16* hl  = hbuf + (l << 13);
            u16*       ml  = mbuf + (l << 13);
            const u16* wx  = wxp  + (size_t)l * (512u * 3584u);
            const u16* wh  = whp  + (size_t)l * (512u * 2048u);
            const u16* wm  = wmp  + (size_t)l * (512u * 1536u);
            const u16* wco = wcop + (size_t)l * (512u * 512u);
            const u16* wmo = wmop + (size_t)l * (512u * 512u);
            const u16* w11 = w11p + (size_t)l * (1024u * 512u);

            v8f ai0  = {0,0,0,0,0,0,0,0}, ai1  = {0,0,0,0,0,0,0,0};
            v8f af0  = {0,0,0,0,0,0,0,0}, af1  = {0,0,0,0,0,0,0,0};
            v8f ag0  = {0,0,0,0,0,0,0,0}, ag1  = {0,0,0,0,0,0,0,0};
            v8f aip0 = {0,0,0,0,0,0,0,0}, aip1 = {0,0,0,0,0,0,0,0};
            v8f afp0 = {0,0,0,0,0,0,0,0}, afp1 = {0,0,0,0,0,0,0,0};
            v8f agp0 = {0,0,0,0,0,0,0,0}, agp1 = {0,0,0,0,0,0,0,0};
            v8f ao0  = {0,0,0,0,0,0,0,0}, ao1  = {0,0,0,0,0,0,0,0};

            // phase 1: gate pre-activations; A frag shared by 14/8/6 WMMAs
            #pragma unroll 1
            for (int kt = 0; kt < 16; ++kt) {
                v16bf a = afrag(inp, kt, lane);
                long fb = (long)kt * 224 + nt0;
                __builtin_prefetch(wx + ((fb + 224) << 9), 0, 1);
                GATE2(ai0,  ai1,  a, wx, fb);
                GATE2(af0,  af1,  a, wx, fb + 32);
                GATE2(ag0,  ag1,  a, wx, fb + 64);
                GATE2(aip0, aip1, a, wx, fb + 96);
                GATE2(afp0, afp1, a, wx, fb + 128);
                GATE2(agp0, agp1, a, wx, fb + 160);
                GATE2(ao0,  ao1,  a, wx, fb + 192);
            }
            #pragma unroll 1
            for (int kt = 0; kt < 16; ++kt) {
                v16bf a = afrag(hl, kt, lane);
                long fb = (long)kt * 128 + nt0;
                GATE2(ai0, ai1, a, wh, fb);
                GATE2(af0, af1, a, wh, fb + 32);
                GATE2(ag0, ag1, a, wh, fb + 64);
                GATE2(ao0, ao1, a, wh, fb + 96);
            }
            #pragma unroll 1
            for (int kt = 0; kt < 16; ++kt) {
                v16bf a = afrag(ml, kt, lane);
                long fb = (long)kt * 96 + nt0;
                GATE2(aip0, aip1, a, wm, fb);
                GATE2(afp0, afp1, a, wm, fb + 32);
                GATE2(agp0, agp1, a, wm, fb + 64);
            }

            // phase 2: element-wise cell updates (C-fragment layout), both tiles
            const float* bl = bvec + l * (7 * HID);
            float bs0[7], bs1[7];
            #pragma unroll
            for (int g = 0; g < 7; ++g) {
                bs0[g] = bl[g * HID + col0];
                bs1[g] = bl[g * HID + col0 + 16];
            }
            v8f mold0 = load_state(ml, nt0, lane);
            v8f mold1 = load_state(ml, nt1, lane);
            v8f cn0, cn1, mn0, mn1;
            #pragma unroll
            for (int r = 0; r < 8; ++r) {
                float iv0  = sigmoidf_(ai0[r]  + bs0[0]);
                float fv0  = sigmoidf_(af0[r]  + bs0[1] + FEPS);
                float gv0  = tanhf_   (ag0[r]  + bs0[2]);
                cn0[r] = fv0 * c_reg[l][0][r] + iv0 * gv0;
                float ipv0 = sigmoidf_(aip0[r] + bs0[3]);
                float fpv0 = sigmoidf_(afp0[r] + bs0[4] + FEPS);
                float gpv0 = tanhf_   (agp0[r] + bs0[5]);
                mn0[r] = fpv0 * mold0[r] + ipv0 * gpv0;

                float iv1  = sigmoidf_(ai1[r]  + bs1[0]);
                float fv1  = sigmoidf_(af1[r]  + bs1[1] + FEPS);
                float gv1  = tanhf_   (ag1[r]  + bs1[2]);
                cn1[r] = fv1 * c_reg[l][1][r] + iv1 * gv1;
                float ipv1 = sigmoidf_(aip1[r] + bs1[3]);
                float fpv1 = sigmoidf_(afp1[r] + bs1[4] + FEPS);
                float gpv1 = tanhf_   (agp1[r] + bs1[5]);
                mn1[r] = fpv1 * mold1[r] + ipv1 * gpv1;
            }
            c_reg[l][0] = cn0;
            c_reg[l][1] = cn1;
            __syncthreads();                       // phase-1 reads of mbuf[l] complete
            store_state(cnew, nt0, lane, cn0);
            store_state(cnew, nt1, lane, cn1);
            store_state(ml,   nt0, lane, mn0);
            store_state(ml,   nt1, lane, mn1);
            __syncthreads();                       // new c/m fragments visible

            // phase 3: o += c_new@Wco + m_new@Wmo ; h_acc = [c;m]@W11
            #pragma unroll 1
            for (int kt = 0; kt < 16; ++kt) {
                v16bf ac = afrag(cnew, kt, lane);
                v16bf am = afrag(ml,   kt, lane);
                long fb = (long)kt * 32 + nt0;
                GATE2(ao0, ao1, ac, wco, fb);
                GATE2(ao0, ao1, am, wmo, fb);
            }
            v8f ah0 = {0,0,0,0,0,0,0,0}, ah1 = {0,0,0,0,0,0,0,0};
            #pragma unroll 1
            for (int kt = 0; kt < 16; ++kt) {
                v16bf ac = afrag(cnew, kt, lane);
                v16bf am = afrag(ml,   kt, lane);
                GATE2(ah0, ah1, ac, w11, (long)kt * 32 + nt0);
                GATE2(ah0, ah1, am, w11, (long)(kt + 16) * 32 + nt0);
            }
            v8f hn0, hn1;
            #pragma unroll
            for (int r = 0; r < 8; ++r) {
                hn0[r] = sigmoidf_(ao0[r] + bs0[6]) * tanhf_(ah0[r]);
                hn1[r] = sigmoidf_(ao1[r] + bs1[6]) * tanhf_(ah1[r]);
            }

            store_state(hbuf + (l << 13), nt0, lane, hn0);
            store_state(hbuf + (l << 13), nt1, lane, hn1);
            if (l == LNUM - 1) {
                u16* ht = hall + ((size_t)(bt * TSEQ + t) << 13);
                store_state(ht, nt0, lane, hn0);
                store_state(ht, nt1, lane, hn1);
            }
            __syncthreads();                       // h_new visible to next layer / step
        }
    }
}

// ---------- output GEMM: [B*T,H] @ [H,O] + b ----------

__global__ void __launch_bounds__(256)
fc_kernel(const u16* __restrict__ hall, const u16* __restrict__ fcp,
          const float* __restrict__ fcb, float* __restrict__ out) {
    const int lane = threadIdx.x & 31;
    const int wid  = (blockIdx.x << 3) + (threadIdx.x >> 5);
    const int nt   = wid & 31;              // O/16 = 32 column tiles
    const int mt   = wid >> 5;              // bt*T + t
    if (mt >= (BSZ / 16) * TSEQ) return;
    const int bt = mt >> 9;
    const int t  = mt & 511;
    const u16* a = hall + ((size_t)mt << 13);
    v8f acc = {0,0,0,0,0,0,0,0};
    #pragma unroll 1
    for (int kt = 0; kt < 16; ++kt) {
        v16bf av = afrag(a, kt, lane);
        v16bf bv = bfrag(fcp, (long)kt * 32 + nt, lane);
        WMMA(acc, av, bv);
    }
    const int cc = lane & 15;
    const int hi = (lane >> 4) << 3;
    const int oc = (nt << 4) + cc;
    const float bb = fcb[oc];
    #pragma unroll
    for (int r = 0; r < 8; ++r) {
        int row = (bt << 4) + r + hi;
        out[(((size_t)row * TSEQ) + t) * ODIM + oc] = acc[r] + bb;
    }
}

// ---------- launch ----------

extern "C" void kernel_launch(void* const* d_in, const int* in_sizes, int n_in,
                              void* d_out, int out_size, void* d_ws, size_t ws_size,
                              hipStream_t stream) {
    const float* x    = (const float*)d_in[0];
    const float* Wx   = (const float*)d_in[1];
    const float* Wh   = (const float*)d_in[2];
    const float* Wm   = (const float*)d_in[3];
    const float* Wco  = (const float*)d_in[4];
    const float* Wmo  = (const float*)d_in[5];
    const float* W11  = (const float*)d_in[6];
    const float* bvec = (const float*)d_in[7];
    const float* fcw  = (const float*)d_in[8];
    const float* fcb  = (const float*)d_in[9];

    u16* ws = (u16*)d_ws;
    size_t off = 0;
    u16* wxp  = ws + off; off += (size_t)LNUM * 512 * 3584;
    u16* whp  = ws + off; off += (size_t)LNUM * 512 * 2048;
    u16* wmp  = ws + off; off += (size_t)LNUM * 512 * 1536;
    u16* wcop = ws + off; off += (size_t)LNUM * 512 * 512;
    u16* wmop = ws + off; off += (size_t)LNUM * 512 * 512;
    u16* w11p = ws + off; off += (size_t)LNUM * 1024 * 512;
    u16* fcp  = ws + off; off += (size_t)512 * 512;
    u16* xp   = ws + off; off += (size_t)BSZ * TSEQ * DIM;
    u16* hall = ws + off; off += (size_t)BSZ * TSEQ * HID;

    auto packw = [&](const float* s, u16* dpt, int K, int N, int nm) {
        long tot = (long)nm * K * N;
        pack_w_kernel<<<(int)((tot + 255) / 256), 256, 0, stream>>>(s, dpt, K, N, tot);
    };
    packw(Wx,  wxp,  512, 3584, LNUM);
    packw(Wh,  whp,  512, 2048, LNUM);
    packw(Wm,  wmp,  512, 1536, LNUM);
    packw(Wco, wcop, 512, 512,  LNUM);
    packw(Wmo, wmop, 512, 512,  LNUM);
    packw(W11, w11p, 1024, 512, LNUM);
    packw(fcw, fcp,  512, 512,  1);

    {
        long tot = (long)BSZ * TSEQ * DIM;
        pack_x_kernel<<<(int)((tot + 255) / 256), 256, 0, stream>>>(x, xp, tot);
    }

    stlstm_kernel<<<BSZ / 16, 512, 0, stream>>>(wxp, whp, wmp, wcop, wmop, w11p,
                                                bvec, xp, hall);

    fc_kernel<<<(BSZ / 16) * TSEQ * (ODIM / 16) / 8, 256, 0, stream>>>(
        hall, fcp, fcb, (float*)d_out);
}
